// InteractiveDualMambaBlock_90872918049559
// MI455X (gfx1250) — compile-verified, matrix operand order checked by reading, not run on hardware
//
#include <hip/hip_runtime.h>
#include <hip/hip_bf16.h>
#include <math.h>

// ---------------- problem constants ----------------
#define D_MODEL 1024
#define D_STATE 16
#define INNER   2048          // EXPAND * D_MODEL
#define DT_RANK 64            // D_MODEL / 16
#define XDBL_N  96            // DT_RANK + 2*D_STATE
#define B_SZ    2
#define SEQ_L   1024
#define M_ROWS  (B_SZ * SEQ_L)   // 2048 token rows

typedef __attribute__((ext_vector_type(16))) __bf16 v16bf;
typedef __attribute__((ext_vector_type(8)))  float  v8f;

// ---------------------------------------------------------------------------
// Fragment loader: 16 fp32 -> 16 bf16 in the ISA's 16-bit A/B layout.
// Caller passes pointer at (row*ld + k + kh); elements 0..7 come from there,
// elements 8..15 from +16 floats (the K+16 half-tile). 16B-aligned -> b128.
// ---------------------------------------------------------------------------
__device__ __forceinline__ v16bf load_frag_bf16(const float* __restrict__ p) {
    const float4* q0 = reinterpret_cast<const float4*>(p);
    const float4* q1 = reinterpret_cast<const float4*>(p + 16);
    float4 a = q0[0], b = q0[1], c = q1[0], d = q1[1];
    v16bf f;
    f[0]=(__bf16)a.x;  f[1]=(__bf16)a.y;  f[2]=(__bf16)a.z;  f[3]=(__bf16)a.w;
    f[4]=(__bf16)b.x;  f[5]=(__bf16)b.y;  f[6]=(__bf16)b.z;  f[7]=(__bf16)b.w;
    f[8]=(__bf16)c.x;  f[9]=(__bf16)c.y;  f[10]=(__bf16)c.z; f[11]=(__bf16)c.w;
    f[12]=(__bf16)d.x; f[13]=(__bf16)d.y; f[14]=(__bf16)d.z; f[15]=(__bf16)d.w;
    return f;
}

// ---------------------------------------------------------------------------
// Generic C[M,N] = A[M,K] @ W[N,K]^T  (row-major, bf16 WMMA, fp32 accum).
// Templated tile counts -> zero in-loop control flow, EXEC always all-1s.
// Block = 256 threads = 8 waves. Wave handles MT m-tiles x NT n-tiles
// (MT*16 rows x NT*16 cols), reusing each B-fragment MT times and each
// A-fragment NT times: MT*NT WMMAs per (MT+NT) fragment loads per K-step.
// Requires: M % (128*MT) == 0, N % (16*NT) == 0, K % 32 == 0.
// ---------------------------------------------------------------------------
template<int NT, int MT>
__global__ void wmma_gemm_bf16(const float* __restrict__ A,
                               const float* __restrict__ W,
                               float* __restrict__ C,
                               int K, int lda, int ldw, int ldc) {
    const int lane  = threadIdx.x & 31;
    const int wave  = threadIdx.x >> 5;
    const int m0    = (blockIdx.y * 8 + wave) * (16 * MT);
    const int nbase = blockIdx.x * (16 * NT);
    const int lr    = lane & 15;          // row (A) / col (B) within tile
    const int kh    = (lane >> 4) << 3;   // lanes 16-31 take the +8 K half

    v8f acc[MT][NT];
    #pragma unroll
    for (int mt = 0; mt < MT; ++mt)
        #pragma unroll
        for (int nt = 0; nt < NT; ++nt)
            acc[mt][nt] = v8f{};

    const float* arow[MT];
    #pragma unroll
    for (int mt = 0; mt < MT; ++mt)
        arow[mt] = A + (size_t)(m0 + mt * 16 + lr) * lda + kh;
    const float* wrow[NT];
    #pragma unroll
    for (int nt = 0; nt < NT; ++nt)
        wrow[nt] = W + (size_t)(nbase + nt * 16 + lr) * ldw + kh;

    for (int k = 0; k < K; k += 32) {
        v16bf af[MT];
        #pragma unroll
        for (int mt = 0; mt < MT; ++mt)
            af[mt] = load_frag_bf16(arow[mt] + k);
        #pragma unroll
        for (int nt = 0; nt < NT; ++nt) {
            v16bf bf = load_frag_bf16(wrow[nt] + k);
            #pragma unroll
            for (int mt = 0; mt < MT; ++mt)
                acc[mt][nt] = __builtin_amdgcn_wmma_f32_16x16x32_bf16(
                    false, af[mt], false, bf, (short)0, acc[mt][nt], false, false);
        }
    }

    // C/D layout: VGPR r -> row tile_m0 + r + 8*(lane>=16), col n0 + (lane&15)
    const int rbias = (lane >> 4) << 3;
    #pragma unroll
    for (int mt = 0; mt < MT; ++mt) {
        const int mrow = m0 + mt * 16 + rbias;
        #pragma unroll
        for (int nt = 0; nt < NT; ++nt) {
            #pragma unroll
            for (int r = 0; r < 8; ++r)
                C[(size_t)(mrow + r) * ldc + nbase + nt * 16 + lr] = acc[mt][nt][r];
        }
    }
}

__device__ __forceinline__ float silu_f(float x) {
    return x / (1.f + __expf(-x));
}
__device__ __forceinline__ float softplus_f(float x) {
    return (x > 20.f) ? x : log1pf(__expf(x));
}

// ---------------------------------------------------------------------------
// Depthwise causal convs (k=2, k=4) + 'same' conv (k=3) + gate:
//   xf = silu(conv1(xi)) * silu(conv2(xi)) + conv_tune(xi)
// xi = left half of xr [M, 4096]; one thread per (m, d), taps along L.
// ---------------------------------------------------------------------------
__global__ void conv_gate_kernel(const float* __restrict__ xr,
                                 const float* __restrict__ w1, const float* __restrict__ b1,
                                 const float* __restrict__ w2, const float* __restrict__ b2,
                                 const float* __restrict__ wt, const float* __restrict__ bt,
                                 float* __restrict__ xf) {
    size_t idx = (size_t)blockIdx.x * blockDim.x + threadIdx.x;
    if (idx >= (size_t)M_ROWS * INNER) return;
    int d = (int)(idx & (INNER - 1));
    size_t m = idx >> 11;                 // b*L + l
    int l = (int)(m & (SEQ_L - 1));
    size_t rowbase = (m - (size_t)l) << 12;   // b*L * 4096

    auto tap = [&](int lp) -> float {
        if (lp < 0 || lp >= SEQ_L) return 0.f;
        return xr[rowbase + ((size_t)lp << 12) + d];
    };
    float xm3 = tap(l - 3), xm2 = tap(l - 2), xm1 = tap(l - 1);
    float x0  = tap(l),     xp1 = tap(l + 1);

    float c1 = b1[d] + w1[2*d]*xm1 + w1[2*d+1]*x0;
    float c2 = b2[d] + w2[4*d]*xm3 + w2[4*d+1]*xm2 + w2[4*d+2]*xm1 + w2[4*d+3]*x0;
    float ct = bt[d] + wt[3*d]*xm1 + wt[3*d+1]*x0  + wt[3*d+2]*xp1;

    xf[m * INNER + d] = silu_f(c1) * silu_f(c2) + ct;
}

// dt = softplus(softplus(dt_raw + bias))  (in-place, [M, INNER])
__global__ void dt_softplus_kernel(float* __restrict__ dt,
                                   const float* __restrict__ bias) {
    size_t idx = (size_t)blockIdx.x * blockDim.x + threadIdx.x;
    if (idx >= (size_t)M_ROWS * INNER) return;
    int d = (int)(idx & (INNER - 1));
    float v = dt[idx] + bias[d];
    dt[idx] = softplus_f(softplus_f(v));
}

// ---------------------------------------------------------------------------
// Selective scan. Lane = state index n (16 per chain), 2 chains per wave32,
// 16 chains per 256-thread block. Sequential over L; per step:
//   h = exp(dt*A)*h + dt*xf*B ;  y = sum_n h*C  (16-lane xor-shuffle reduce)
// Fused epilogue: y = (y + D*xf) * silu(res), written to yg.
// ---------------------------------------------------------------------------
__global__ void scan_kernel(const float* __restrict__ dt,
                            const float* __restrict__ xf,
                            const float* __restrict__ xdbl,
                            const float* __restrict__ A_log,
                            const float* __restrict__ Dv,
                            const float* __restrict__ xr,   // res = xr[:, INNER:]
                            float* __restrict__ yg) {
    int chain = blockIdx.x * (blockDim.x >> 4) + (threadIdx.x >> 4);
    int n = threadIdx.x & 15;
    int b = chain >> 11;             // chain / INNER
    int d = chain & (INNER - 1);

    float A_dn = -__expf(A_log[(size_t)d * D_STATE + n]);
    float Dd   = Dv[d];
    float h    = 0.f;

    size_t row = (size_t)b * SEQ_L;
    for (int t = 0; t < SEQ_L; ++t, ++row) {
        float dtv = dt[row * INNER + d];
        float xfv = xf[row * INNER + d];
        float Bv  = xdbl[row * XDBL_N + DT_RANK + n];
        float Cv  = xdbl[row * XDBL_N + DT_RANK + D_STATE + n];

        h = __expf(dtv * A_dn) * h + dtv * xfv * Bv;

        float p = h * Cv;
        p += __shfl_xor(p, 8, 16);
        p += __shfl_xor(p, 4, 16);
        p += __shfl_xor(p, 2, 16);
        p += __shfl_xor(p, 1, 16);

        if (n == 0) {
            float rv = xr[(row << 12) + INNER + d];
            yg[row * INNER + d] = (p + Dd * xfv) * silu_f(rv);
        }
    }
}

// ---------------------------------------------------------------------------
extern "C" void kernel_launch(void* const* d_in, const int* in_sizes, int n_in,
                              void* d_out, int out_size, void* d_ws, size_t ws_size,
                              hipStream_t stream) {
    const float* x         = (const float*)d_in[0];
    const float* in_proj_w = (const float*)d_in[1];
    const float* conv1_w   = (const float*)d_in[2];
    const float* conv1_b   = (const float*)d_in[3];
    const float* conv2_w   = (const float*)d_in[4];
    const float* conv2_b   = (const float*)d_in[5];
    const float* tune_w    = (const float*)d_in[6];
    const float* tune_b    = (const float*)d_in[7];
    const float* x_proj_w  = (const float*)d_in[8];
    const float* dt_proj_w = (const float*)d_in[9];
    const float* dt_proj_b = (const float*)d_in[10];
    const float* A_log     = (const float*)d_in[11];
    const float* Dvec      = (const float*)d_in[12];
    const float* out_proj_w= (const float*)d_in[13];
    float* out = (float*)d_out;

    // workspace layout (floats)
    float* xr   = (float*)d_ws;                               // [2048, 4096]
    float* xf   = xr   + (size_t)M_ROWS * (2 * INNER);        // [2048, 2048]
    float* xdbl = xf   + (size_t)M_ROWS * INNER;              // [2048, 96]
    float* dt   = xdbl + (size_t)M_ROWS * XDBL_N;             // [2048, 2048]
    float* yg   = dt   + (size_t)M_ROWS * INNER;              // [2048, 2048]

    const dim3 blk(256);

    // 1) xr = x @ in_proj_w^T   M=2048, N=4096, K=1024  (NT=4, MT=2)
    hipLaunchKernelGGL((wmma_gemm_bf16<4,2>),
                       dim3(4096 / 64, M_ROWS / 256), blk, 0, stream,
                       x, in_proj_w, xr, D_MODEL, D_MODEL, D_MODEL, 2 * INNER);

    // 2) depthwise convs + gate -> xf
    {
        size_t total = (size_t)M_ROWS * INNER;
        hipLaunchKernelGGL(conv_gate_kernel, dim3((total + 255) / 256), blk, 0, stream,
                           xr, conv1_w, conv1_b, conv2_w, conv2_b, tune_w, tune_b, xf);
    }

    // 3) x_dbl = xf @ x_proj_w^T   M=2048, N=96, K=2048  (NT=6, MT=1)
    hipLaunchKernelGGL((wmma_gemm_bf16<6,1>),
                       dim3(XDBL_N / 96, M_ROWS / 128), blk, 0, stream,
                       xf, x_proj_w, xdbl, INNER, INNER, INNER, XDBL_N);

    // 4) dt_raw = x_dbl[:, :64] @ dt_proj_w^T   M=2048, N=2048, K=64 (lda=96)
    hipLaunchKernelGGL((wmma_gemm_bf16<4,2>),
                       dim3(INNER / 64, M_ROWS / 256), blk, 0, stream,
                       xdbl, dt_proj_w, dt, DT_RANK, XDBL_N, DT_RANK, INNER);

    // 5) dt = softplus(softplus(dt_raw + bias))
    {
        size_t total = (size_t)M_ROWS * INNER;
        hipLaunchKernelGGL(dt_softplus_kernel, dim3((total + 255) / 256), blk, 0, stream,
                           dt, dt_proj_b);
    }

    // 6) selective scan + gating -> yg
    {
        int chains = B_SZ * INNER;                 // 4096
        hipLaunchKernelGGL(scan_kernel, dim3(chains / 16), blk, 0, stream,
                           dt, xf, xdbl, A_log, Dvec, xr, yg);
    }

    // 7) out = yg @ out_proj_w^T   M=2048, N=1024, K=2048  (NT=4, MT=2)
    hipLaunchKernelGGL((wmma_gemm_bf16<4,2>),
                       dim3(D_MODEL / 64, M_ROWS / 256), blk, 0, stream,
                       yg, out_proj_w, out, INNER, INNER, INNER, D_MODEL);
}
